// GTTP_25855703122413
// MI455X (gfx1250) — matrix operand order repaired and checked by reading, not compiled, not run on hardware
//
#include <hip/hip_runtime.h>
#include <hip/hip_bf16.h>

typedef __attribute__((ext_vector_type(16))) _Float16 v16h;
typedef __attribute__((ext_vector_type(8)))  _Float16 v8h;
typedef __attribute__((ext_vector_type(8)))  float    v8f;

#define GN   20000
#define GE   320000
#define GB   4096
#define GD   512

// ---------------------------------------------------------------------------
// helpers
// ---------------------------------------------------------------------------
__device__ __forceinline__ unsigned f2mono(float f) {
    unsigned u = __float_as_uint(f);
    return (u & 0x80000000u) ? ~u : (u | 0x80000000u);
}
__device__ __forceinline__ float mono2f(unsigned u) {
    unsigned b = (u & 0x80000000u) ? (u & 0x7fffffffu) : ~u;
    return __uint_as_float(b);
}
__device__ __forceinline__ float wave_sum(float v) {
    #pragma unroll
    for (int off = 16; off; off >>= 1) v += __shfl_xor(v, off, 32);
    return v;
}
// CDNA5 async memory->LDS copy (ASYNCcnt), bypasses VGPRs.
// LDS address = low 32 bits of generic pointer (ISA aperture mapping).
__device__ __forceinline__ void async_copy_b128(unsigned lds, const void* g) {
    asm volatile("global_load_async_to_lds_b128 %0, %1, off"
                 :: "v"(lds), "v"((unsigned long long)(size_t)g)
                 : "memory");
}
__device__ __forceinline__ void wait_async0() {
    asm volatile("s_wait_asynccnt 0x0" ::: "memory");
}
__device__ __forceinline__ unsigned lds_off(const void* p) {
    return (unsigned)(size_t)p;
}

// ---------------------------------------------------------------------------
// conversion / fill kernels
// ---------------------------------------------------------------------------
__global__ __launch_bounds__(256) void k_fill_f32(float* p, float v, int n) {
    int i = blockIdx.x * 256 + threadIdx.x;
    if (i < n) p[i] = v;
}
__global__ __launch_bounds__(256) void k_fill_u32(unsigned* p, unsigned v, int n) {
    int i = blockIdx.x * 256 + threadIdx.x;
    if (i < n) p[i] = v;
}
// W f32 [K, Nc] row-major  ->  Wt f16 [Nc, Kp] (zero-padded K)
__global__ __launch_bounds__(256) void k_convert_wt(const float* __restrict__ W,
                                                    _Float16* __restrict__ Wt,
                                                    int K, int Kp, int Nc) {
    int i = blockIdx.x * 256 + threadIdx.x;
    int tot = Nc * Kp;
    if (i >= tot) return;
    int n = i / Kp, kk = i - n * Kp;
    float v = (kk < K) ? W[(size_t)kk * Nc + n] : 0.f;
    Wt[i] = (_Float16)v;
}
// x f32 [N,102] -> xpad f16 [N,128]
__global__ __launch_bounds__(256) void k_convert_xpad(const float* __restrict__ x,
                                                      _Float16* __restrict__ xp,
                                                      int N, int Kin, int Kp) {
    int i = blockIdx.x * 256 + threadIdx.x;
    if (i >= N * Kp) return;
    int r = i / Kp, c = i - r * Kp;
    float v = (c < Kin) ? x[(size_t)r * Kin + c] : 0.f;
    xp[i] = (_Float16)v;
}

// ---------------------------------------------------------------------------
// WMMA GEMM:  C[M,Nc] = A_f16[M,Kp] * Wt_f16[Nc,Kp]^T + bias   (opt relu)
// Block tile 128x64, BK=32, 8 waves -> each wave owns 32x32 (2x2 wmma tiles).
// Double-buffered LDS fed by global_load_async_to_lds_b128 (ASYNCcnt):
// next K-tile's copies overlap current K-tile's WMMAs; 1 barrier / iter.
// ---------------------------------------------------------------------------
#define BM 128
#define BN 64
#define BK 32
#define APAD 8

__global__ __launch_bounds__(256) void k_wmma_gemm(
    const _Float16* __restrict__ A, int Kp,
    const _Float16* __restrict__ Wt,
    const float* __restrict__ bias,
    float* __restrict__ Cf, _Float16* __restrict__ Ch,
    int M, int Nc, int relu)
{
    __shared__ _Float16 As[2][BM][BK + APAD];
    __shared__ _Float16 Bs[2][BN][BK + APAD];

    const int tid  = threadIdx.x;
    const int lane = tid & 31;
    const int wave = tid >> 5;
    const int wr   = wave & 3;   // row quadrant (0..3) of 32 rows
    const int wc   = wave >> 2;  // col half    (0..1) of 32 cols
    const int row0 = blockIdx.x * BM;
    const int col0 = blockIdx.y * BN;

    const int m  = lane & 15;
    const int kh = lane >> 4;    // 0/1 half-group per ISA layout

    // staging assignment: A tile 128x32 halves -> 32B/thread (2x b128),
    //                     B tile  64x32 halves -> 16B/thread (1x b128)
    const int ra   = tid >> 1, sega = tid & 1;
    const int nb   = tid & 63, segb = tid >> 6;
    int ga_row = row0 + ra; if (ga_row >= M) ga_row = M - 1;  // dup row; masked at store

    v8f acc[2][2] = {};

    auto stage = [&](int kt, int buf) {
        const _Float16* gA = &A[(size_t)ga_row * Kp + kt + sega * 16];
        unsigned lA = lds_off(&As[buf][ra][sega * 16]);
        async_copy_b128(lA,      gA);
        async_copy_b128(lA + 16, gA + 8);
        const _Float16* gB = &Wt[(size_t)(col0 + nb) * Kp + kt + segb * 8];
        async_copy_b128(lds_off(&Bs[buf][nb][segb * 8]), gB);
    };

    stage(0, 0);
    int cur = 0;
    for (int kt = 0; kt < Kp; kt += BK) {
        wait_async0();       // this wave's copies into As/Bs[cur] landed in LDS
        __syncthreads();     // everyone's copies landed; prev-iter reads retired
        if (kt + BK < Kp) stage(kt + BK, cur ^ 1);   // prefetch next tile

        v16h af[2], bf[2];
        #pragma unroll
        for (int i = 0; i < 2; ++i) {
            // A 16x32 layout: lane m rows; halves 0..7 = K kh*8..+7, 8..15 = K 16+kh*8..+7
            const _Float16* p = &As[cur][wr * 32 + i * 16 + m][kh * 8];
            v8h lo = *(const v8h*)p;
            v8h hi = *(const v8h*)(p + 16);
            #pragma unroll
            for (int t = 0; t < 8; ++t) { af[i][t] = lo[t]; af[i][8 + t] = hi[t]; }
        }
        #pragma unroll
        for (int j = 0; j < 2; ++j) {
            // B 32x16 layout: col = lane%16; halves 0..7 = K kh*8..+7, 8..15 = K 16+kh*8..+7
            const _Float16* p = &Bs[cur][wc * 32 + j * 16 + m][kh * 8];
            v8h lo = *(const v8h*)p;
            v8h hi = *(const v8h*)(p + 16);
            #pragma unroll
            for (int t = 0; t < 8; ++t) { bf[j][t] = lo[t]; bf[j][8 + t] = hi[t]; }
        }
        #pragma unroll
        for (int i = 0; i < 2; ++i)
            #pragma unroll
            for (int j = 0; j < 2; ++j)
                acc[i][j] = __builtin_amdgcn_wmma_f32_16x16x32_f16(
                    false, af[i], false, bf[j], (short)0, acc[i][j], false, false);
        cur ^= 1;
    }

    // epilogue: C/D layout — lane l, vgpr r -> (row r+8*(l>>4), col l&15)
    #pragma unroll
    for (int i = 0; i < 2; ++i) {
        #pragma unroll
        for (int j = 0; j < 2; ++j) {
            int tr = row0 + wr * 32 + i * 16 + 8 * kh;
            int tc = col0 + wc * 32 + j * 16 + m;
            float bv = bias ? bias[tc] : 0.f;
            #pragma unroll
            for (int r = 0; r < 8; ++r) {
                int rr = tr + r;
                if (rr < M) {
                    float v = acc[i][j][r] + bv;
                    if (relu) v = fmaxf(v, 0.f);
                    if (Cf) Cf[(size_t)rr * Nc + tc] = v;
                    if (Ch) Ch[(size_t)rr * Nc + tc] = (_Float16)v;
                }
            }
        }
    }
}

// ---------------------------------------------------------------------------
// edge kernels (segment softmax attention), wave-per-edge
// ---------------------------------------------------------------------------
__global__ __launch_bounds__(256) void k_edge_score(
    const float* __restrict__ q, const float* __restrict__ k,
    const float* __restrict__ we, const float* __restrict__ be,
    const float* __restrict__ ea, const int* __restrict__ src,
    const int* __restrict__ dst, float* __restrict__ score, int E)
{
    int e = blockIdx.x * 8 + (threadIdx.x >> 5);
    if (e >= E) return;
    int lane = threadIdx.x & 31;
    int s = src[e], d = dst[e];
    float a = ea[e];
    const float* qr = q + (size_t)d * GD;
    const float* kr = k + (size_t)s * GD;
    float acc = 0.f;
    for (int c = lane; c < GD; c += 32)
        acc = fmaf(qr[c], kr[c] + fmaf(a, we[c], be[c]), acc);
    acc = wave_sum(acc);
    if (lane == 0) score[e] = acc * 0.04419417382415922f;  // 1/sqrt(512)
}

__global__ __launch_bounds__(256) void k_seg_max(
    const float* __restrict__ score, const int* __restrict__ dst,
    unsigned* __restrict__ m_u, int E)
{
    int e = blockIdx.x * 256 + threadIdx.x;
    if (e >= E) return;
    atomicMax(&m_u[dst[e]], f2mono(score[e]));
}

__global__ __launch_bounds__(256) void k_edge_exp(
    const float* __restrict__ score, const int* __restrict__ dst,
    const unsigned* __restrict__ m_u, float* __restrict__ ex,
    float* __restrict__ denom, int E)
{
    int e = blockIdx.x * 256 + threadIdx.x;
    if (e >= E) return;
    unsigned mu = m_u[dst[e]];
    float m = (mu == 0u) ? 0.f : mono2f(mu);
    if (!isfinite(m)) m = 0.f;
    float v = __expf(score[e] - m);
    ex[e] = v;
    atomicAdd(&denom[dst[e]], v);
}

__global__ __launch_bounds__(256) void k_edge_scatter(
    const float* __restrict__ vmat, const float* __restrict__ we,
    const float* __restrict__ be, const float* __restrict__ ea,
    const float* __restrict__ ex, const float* __restrict__ denom,
    const int* __restrict__ src, const int* __restrict__ dst,
    float* __restrict__ agg, int E)
{
    int e = blockIdx.x * 8 + (threadIdx.x >> 5);
    if (e >= E) return;
    int lane = threadIdx.x & 31;
    int s = src[e], d = dst[e];
    float dn = denom[d];
    float alpha = ex[e] / (dn > 0.f ? dn : 1.f);
    float a = ea[e];
    const float* vr = vmat + (size_t)s * GD;
    float* ar = agg + (size_t)d * GD;
    for (int c = lane; c < GD; c += 32)
        atomicAdd(&ar[c], alpha * (vr[c] + fmaf(a, we[c], be[c])));
}

// beta gate + combine + (relu) + f16 for next layer's WMMA A operand
__global__ __launch_bounds__(256) void k_beta(
    const float* __restrict__ skip, const float* __restrict__ agg,
    const float* __restrict__ bw, _Float16* __restrict__ h16,
    int relu, int N)
{
    int n = blockIdx.x * 8 + (threadIdx.x >> 5);
    if (n >= N) return;
    int lane = threadIdx.x & 31;
    const float* sr = skip + (size_t)n * GD;
    const float* ar = agg + (size_t)n * GD;
    float acc = 0.f;
    for (int c = lane; c < GD; c += 32) {
        float sv = sr[c], av = ar[c];
        acc += sv * bw[c] + av * bw[GD + c] + (sv - av) * bw[2 * GD + c];
    }
    acc = wave_sum(acc);
    float beta = 1.f / (1.f + __expf(-acc));
    for (int c = lane; c < GD; c += 32) {
        float v = beta * sr[c] + (1.f - beta) * ar[c];
        if (relu) v = fmaxf(v, 0.f);
        h16[(size_t)n * GD + c] = (_Float16)v;
    }
}

// ---------------------------------------------------------------------------
// head: gather [s|t|a|b] feats (f16), LN, per-row dot
// ---------------------------------------------------------------------------
__global__ __launch_bounds__(256) void k_gather_feats(
    const _Float16* __restrict__ h16, const int* __restrict__ si,
    const int* __restrict__ ti, const int* __restrict__ ai,
    const int* __restrict__ bi, _Float16* __restrict__ feats)
{
    int row = blockIdx.x;
    int n0 = si[row], n1 = ti[row], n2 = ai[row], n3 = bi[row];
    for (int c = threadIdx.x; c < 4 * GD; c += 256) {
        int which = c >> 9, d = c & (GD - 1);
        int node = (which == 0) ? n0 : (which == 1) ? n1 : (which == 2) ? n2 : n3;
        feats[(size_t)row * 4 * GD + c] = h16[(size_t)node * GD + d];
    }
}

__global__ __launch_bounds__(256) void k_layernorm(
    const float* __restrict__ x, const float* __restrict__ g,
    const float* __restrict__ b, _Float16* __restrict__ y, int Dn)
{
    int row = blockIdx.x;
    const float* xr = x + (size_t)row * Dn;
    __shared__ float rs[8], rs2[8];
    float s = 0.f, s2 = 0.f;
    for (int c = threadIdx.x; c < Dn; c += 256) {
        float v = xr[c]; s += v; s2 += v * v;
    }
    s = wave_sum(s); s2 = wave_sum(s2);
    if ((threadIdx.x & 31) == 0) { rs[threadIdx.x >> 5] = s; rs2[threadIdx.x >> 5] = s2; }
    __syncthreads();
    float ts = 0.f, ts2 = 0.f;
    #pragma unroll
    for (int i = 0; i < 8; ++i) { ts += rs[i]; ts2 += rs2[i]; }
    float mu  = ts / Dn;
    float var = ts2 / Dn - mu * mu;
    float inv = rsqrtf(var + 1e-5f);
    for (int c = threadIdx.x; c < Dn; c += 256)
        y[(size_t)row * Dn + c] = (_Float16)((xr[c] - mu) * inv * g[c] + b[c]);
}

__global__ __launch_bounds__(256) void k_head(
    const _Float16* __restrict__ h2, const float* __restrict__ hw,
    const float* __restrict__ hb, float* __restrict__ out, int Brows)
{
    int row = blockIdx.x * 8 + (threadIdx.x >> 5);
    if (row >= Brows) return;
    int lane = threadIdx.x & 31;
    const _Float16* hr = h2 + (size_t)row * GD;
    float acc = 0.f;
    for (int c = lane; c < GD; c += 32)
        acc = fmaf((float)hr[c], hw[c], acc);
    acc = wave_sum(acc);
    if (lane == 0) out[row] = acc + hb[0];
}

// ---------------------------------------------------------------------------
// host launcher
// ---------------------------------------------------------------------------
static inline dim3 g1(int n, int t) { return dim3((n + t - 1) / t); }

extern "C" void kernel_launch(void* const* d_in, const int* in_sizes, int n_in,
                              void* d_out, int out_size, void* d_ws, size_t ws_size,
                              hipStream_t stream) {
    (void)in_sizes; (void)n_in; (void)out_size; (void)ws_size;

    // ---- inputs (setup_inputs dict insertion order; params leaves in nested
    //      dict insertion order: per conv {q.w,q.b,k.w,k.b,v.w,v.b,e.w,e.b,
    //      skip.w,skip.b,beta_w} x3, then mlp1.w,mlp1.b,ln_g,ln_b,mlp2.w,
    //      mlp2.b,head.w,head.b) ----
    const float* x   = (const float*)d_in[0];
    const int* ei    = (const int*)d_in[1];
    const float* ea  = (const float*)d_in[2];
    const int* s_idx = (const int*)d_in[3];
    const int* e_idx = (const int*)d_in[4];
    const int* x1i   = (const int*)d_in[5];
    const int* x2i   = (const int*)d_in[6];
    const int* src = ei;          // edge_index[0]
    const int* dst = ei + GE;     // edge_index[1]

    const float* cw[3][11];
    for (int l = 0; l < 3; ++l)
        for (int j = 0; j < 11; ++j)
            cw[l][j] = (const float*)d_in[7 + l * 11 + j];
    // per conv: 0 qw 1 qb 2 kw 3 kb 4 vw 5 vb 6 ew 7 eb 8 skw 9 skb 10 betaw
    const float* mlp1w = (const float*)d_in[40];
    const float* mlp1b = (const float*)d_in[41];
    const float* ln_g  = (const float*)d_in[42];
    const float* ln_b  = (const float*)d_in[43];
    const float* mlp2w = (const float*)d_in[44];
    const float* mlp2b = (const float*)d_in[45];
    const float* headw = (const float*)d_in[46];
    const float* headb = (const float*)d_in[47];

    // ---- workspace bump allocator ----
    char* base = (char*)d_ws;
    size_t off = 0;
    auto alloc = [&](size_t bytes) -> void* {
        void* p = base + off;
        off = (off + bytes + 255) & ~(size_t)255;
        return p;
    };
    _Float16* xpad = (_Float16*)alloc((size_t)GN * 128 * 2);
    _Float16* h16  = (_Float16*)alloc((size_t)GN * GD * 2);
    float* q32 = (float*)alloc((size_t)GN * GD * 4);
    float* k32 = (float*)alloc((size_t)GN * GD * 4);
    float* v32 = (float*)alloc((size_t)GN * GD * 4);
    float* s32 = (float*)alloc((size_t)GN * GD * 4);
    float* agg = (float*)alloc((size_t)GN * GD * 4);
    float* score = (float*)alloc((size_t)GE * 4);
    float* exb   = (float*)alloc((size_t)GE * 4);
    unsigned* m_u = (unsigned*)alloc((size_t)GN * 4);
    float* denom  = (float*)alloc((size_t)GN * 4);
    _Float16* wt[3][4];                 // q,k,v,skip transposed f16 weights
    int kp_l[3] = {128, GD, GD};
    for (int l = 0; l < 3; ++l)
        for (int j = 0; j < 4; ++j)
            wt[l][j] = (_Float16*)alloc((size_t)GD * kp_l[l] * 2);
    _Float16* wt1 = (_Float16*)alloc((size_t)(2 * GD) * (4 * GD) * 2);  // [1024][2048]
    _Float16* wt2 = (_Float16*)alloc((size_t)GD * (2 * GD) * 2);        // [512][1024]
    _Float16* feats = (_Float16*)alloc((size_t)GB * 4 * GD * 2);
    float* h1      = (float*)alloc((size_t)GB * 2 * GD * 4);
    _Float16* h1n  = (_Float16*)alloc((size_t)GB * 2 * GD * 2);
    _Float16* h2   = (_Float16*)alloc((size_t)GB * GD * 2);

    // ---- weight conversions (f32 [K,Nc] -> f16 transposed [Nc,Kp]) ----
    int kin_l[3] = {102, GD, GD};
    int widx[4] = {0, 2, 4, 8};  // q.w, k.w, v.w, skip.w
    for (int l = 0; l < 3; ++l)
        for (int j = 0; j < 4; ++j) {
            int tot = GD * kp_l[l];
            k_convert_wt<<<g1(tot, 256), 256, 0, stream>>>(
                cw[l][widx[j]], wt[l][j], kin_l[l], kp_l[l], GD);
        }
    k_convert_wt<<<g1(2 * GD * 4 * GD, 256), 256, 0, stream>>>(
        mlp1w, wt1, 4 * GD, 4 * GD, 2 * GD);
    k_convert_wt<<<g1(GD * 2 * GD, 256), 256, 0, stream>>>(
        mlp2w, wt2, 2 * GD, 2 * GD, GD);
    k_convert_xpad<<<g1(GN * 128, 256), 256, 0, stream>>>(x, xpad, GN, 102, 128);

    // ---- 3 TransformerConv layers ----
    const _Float16* hcur = xpad;
    int kp = 128;
    for (int l = 0; l < 3; ++l) {
        dim3 gg(g1(GN, BM).x, GD / BN);
        float* outs[4] = {q32, k32, v32, s32};
        int bidx[4] = {1, 3, 5, 9};  // q.b, k.b, v.b, skip.b
        for (int j = 0; j < 4; ++j)
            k_wmma_gemm<<<gg, 256, 0, stream>>>(
                hcur, kp, wt[l][j], cw[l][bidx[j]], outs[j], (_Float16*)nullptr,
                GN, GD, 0);

        k_fill_u32<<<g1(GN, 256), 256, 0, stream>>>(m_u, 0u, GN);
        k_fill_f32<<<g1(GN, 256), 256, 0, stream>>>(denom, 0.f, GN);
        k_fill_f32<<<g1(GN * GD, 256), 256, 0, stream>>>(agg, 0.f, GN * GD);

        const float* we = cw[l][6];  // e.w [1,512]
        const float* be = cw[l][7];  // e.b
        k_edge_score<<<g1(GE, 8), 256, 0, stream>>>(q32, k32, we, be, ea, src, dst,
                                                    score, GE);
        k_seg_max<<<g1(GE, 256), 256, 0, stream>>>(score, dst, m_u, GE);
        k_edge_exp<<<g1(GE, 256), 256, 0, stream>>>(score, dst, m_u, exb, denom, GE);
        k_edge_scatter<<<g1(GE, 8), 256, 0, stream>>>(v32, we, be, ea, exb, denom,
                                                      src, dst, agg, GE);
        k_beta<<<g1(GN, 8), 256, 0, stream>>>(s32, agg, cw[l][10], h16,
                                              (l < 2) ? 1 : 0, GN);
        hcur = h16;
        kp = GD;
    }

    // ---- head MLP ----
    k_gather_feats<<<GB, 256, 0, stream>>>(h16, s_idx, e_idx, x1i, x2i, feats);
    {   // mlp1: [4096,2048] x [2048,1024] + relu -> f32
        dim3 gg(g1(GB, BM).x, (2 * GD) / BN);
        k_wmma_gemm<<<gg, 256, 0, stream>>>(feats, 4 * GD, wt1, mlp1b,
                                            h1, (_Float16*)nullptr, GB, 2 * GD, 1);
    }
    k_layernorm<<<GB, 256, 0, stream>>>(h1, ln_g, ln_b, h1n, 2 * GD);
    {   // mlp2: [4096,1024] x [1024,512] + relu -> f16
        dim3 gg(g1(GB, BM).x, GD / BN);
        k_wmma_gemm<<<gg, 256, 0, stream>>>(h1n, 2 * GD, wt2, mlp2b,
                                            (float*)nullptr, h2, GB, GD, 1);
    }
    k_head<<<g1(GB, 8), 256, 0, stream>>>(h2, headw, headb, (float*)d_out, GB);
}
